// SALayer_9723805958285
// MI455X (gfx1250) — matrix operand order repaired and checked by reading, not compiled, not run on hardware
//
#include <hip/hip_runtime.h>
#include <hip/hip_bf16.h>

#define HID   128
#define NHEAD 8

typedef __attribute__((ext_vector_type(16))) _Float16 v16h;
typedef __attribute__((ext_vector_type(8)))  float    v8f;

__device__ __forceinline__ v8f wmma16(v16h a, v16h b, v8f c) {
  // D = A(16x32) x B(32x16) + C(16x16 f32)
  return __builtin_amdgcn_wmma_f32_16x16x32_f16(false, a, false, b, (short)0, c, false, false);
}

// k index (even) held by dword slot i of an A fragment, for lane-half h
__device__ __forceinline__ int a_kmap(int i, int half) {
  return 2 * i + 16 * half;
}

// ---- CDNA5 async global->LDS copy (ASYNCcnt path, ISA ch.10 / 08_async_tensor) ----
__device__ __forceinline__ void async_ld_b128(void* lds_dst, const void* gsrc) {
  // generic LDS pointer: low 32 bits are the LDS byte offset
  uint32_t l = (uint32_t)(uintptr_t)lds_dst;
  asm volatile("global_load_async_to_lds_b128 %0, %1, off"
               :: "v"(l), "v"(gsrc) : "memory");
}
__device__ __forceinline__ void wait_async0() {
  asm volatile("s_wait_asynccnt 0x0" ::: "memory");
}

// ---------------------------------------------------------------------------
// Prepack f32 weight [K x N] row-major -> f16 B-fragment order:
// dword d = ((nt*(K/32)+kt)*32 + lane)*8 + i holds (W[k][n], W[k+1][n])
// with n = nt*16 + lane%16, k = kt*32 + 2*i + 16*(lane/16)
// ---------------------------------------------------------------------------
__global__ void prepack_kernel(const float* __restrict__ W, _Float16* __restrict__ Wp,
                               int K, int N) {
  int total = (K * N) >> 1;  // dwords
  int KT = K >> 5;
  for (int d = blockIdx.x * blockDim.x + threadIdx.x; d < total;
       d += gridDim.x * blockDim.x) {
    int i    = d & 7;
    int lane = (d >> 3) & 31;
    int kt   = (d >> 8) % KT;
    int nt   = d / (KT << 8);
    int n = nt * 16 + (lane & 15);
    int k = kt * 32 + 2 * i + ((lane >> 4) << 4);
    Wp[2 * d]     = (_Float16)W[(size_t)k * N + n];
    Wp[2 * d + 1] = (_Float16)W[(size_t)(k + 1) * N + n];
  }
}

// ---------------------------------------------------------------------------
// Generic GEMM: C[M x NOUT] = A[M x KDIM] @ W + bias, optional relu.
// 256 threads = 8 waves, one 16-row tile per wave. Weights AND the per-wave
// A tile are staged into LDS with async b128 copies (no VGPR round-trip).
// In-place (A == C) is safe: tile fully staged in LDS before any store.
// ---------------------------------------------------------------------------
template <int KDIM, int NOUT, bool HASBIAS, bool RELU>
__global__ void gemm_kernel(const float* A, const _Float16* __restrict__ Wp,
                            const float* __restrict__ bias, float* C, int numTiles) {
  extern __shared__ char smem[];
  _Float16* wl = (_Float16*)smem;                                        // KDIM*NOUT f16
  float*    bl = (float*)(smem + (size_t)KDIM * NOUT * sizeof(_Float16)); // NOUT f32
  float* tiles = bl + NOUT;                                               // 8 x (16*KDIM) f32
  const int tid = threadIdx.x, wave = tid >> 5, lane = tid & 31;
  float* atile = tiles + wave * (16 * KDIM);
  const int tile = blockIdx.x * (blockDim.x >> 5) + wave;

  // async-stage weights (whole block) + this wave's A tile
  {
    const int wChunks = (KDIM * NOUT * 2) / 16;
    for (int i = tid; i < wChunks; i += blockDim.x)
      async_ld_b128((char*)wl + i * 16, (const char*)Wp + i * 16);
    if (HASBIAS)
      for (int i = tid; i < NOUT; i += blockDim.x) bl[i] = bias[i];
    if (tile < numTiles) {
      const char* g = (const char*)(A + (size_t)tile * 16 * KDIM);  // 16 rows contiguous
      char*       l = (char*)atile;
      for (int i = lane; i < 4 * KDIM; i += 32)                     // 16B chunks
        async_ld_b128(l + i * 16, g + i * 16);
    }
  }
  wait_async0();
  __syncthreads();
  if (tile >= numTiles) return;

  const int m = lane & 15, half = lane >> 4;
  constexpr int KT = KDIM / 32;

  // A fragments (contiguous k = 16*half + s, same order as prepacked B) from LDS tile
  v16h a[KT];
#pragma unroll
  for (int kt = 0; kt < KT; ++kt) {
#pragma unroll
    for (int i = 0; i < 8; ++i) {
      int k = kt * 32 + a_kmap(i, half);
      a[kt][2 * i]     = (_Float16)atile[m * KDIM + k];
      a[kt][2 * i + 1] = (_Float16)atile[m * KDIM + k + 1];
    }
  }

#pragma unroll
  for (int nt = 0; nt < NOUT / 16; ++nt) {
    v16h bf[KT];
#pragma unroll
    for (int kt = 0; kt < KT; ++kt)
      __builtin_memcpy(&bf[kt], wl + ((((nt * KT + kt) * 32) + lane) << 4), sizeof(v16h));
    v8f acc = {};
#pragma unroll
    for (int kt = 0; kt < KT; ++kt) acc = wmma16(a[kt], bf[kt], acc);

    const int n  = nt * 16 + m;
    const float bv = HASBIAS ? bl[n] : 0.0f;
#pragma unroll
    for (int j = 0; j < 8; ++j) {  // C layout: lane col = n, row = j + 8*half
      float v = acc[j] + bv;
      if (RELU) v = fmaxf(v, 0.0f);
      C[(size_t)(tile * 16 + j + 8 * half) * NOUT + n] = v;
    }
  }
}

// ---------------------------------------------------------------------------
// Edge attention: score = K[src]*Q[dst]*0.25*pe (written back over pe as
// e_attn), s = exp(clip(head-sum)), atomically accumulate V[src]*s -> wV[dst]
// and s -> z[dst]. One wave per edge; lane l owns elements 4l..4l+3 (head l/4).
// ---------------------------------------------------------------------------
__global__ void edge_attn_kernel(const float* __restrict__ Q, const float* __restrict__ Kf,
                                 const float* __restrict__ V, float* __restrict__ e_attn,
                                 const int* __restrict__ src, const int* __restrict__ dst,
                                 float* __restrict__ wV, float* __restrict__ z, int E) {
  const int lane = threadIdx.x & 31;
  const int wave = threadIdx.x >> 5;
  const int wpg  = (blockDim.x >> 5) * gridDim.x;
  const int c    = lane * 4;
  for (int e = blockIdx.x * (blockDim.x >> 5) + wave; e < E; e += wpg) {
    const int si = src[e], di = dst[e];
    const float* kr = Kf + (size_t)si * HID;
    const float* qr = Q + (size_t)di * HID;
    const float* vr = V + (size_t)si * HID;
    float* er = e_attn + (size_t)e * HID;
    float hs = 0.0f;
#pragma unroll
    for (int j = 0; j < 4; ++j) {
      float v = kr[c + j] * qr[c + j] * 0.25f * er[c + j];  // er holds pe on entry
      er[c + j] = v;                                        // e_attn in place
      hs += v;
    }
    hs += __shfl_xor(hs, 1, 32);   // reduce within 4-lane head group
    hs += __shfl_xor(hs, 2, 32);
    const float sexp = expf(fminf(fmaxf(hs, -5.0f), 5.0f));
#pragma unroll
    for (int j = 0; j < 4; ++j)
      atomicAdd(&wV[(size_t)di * HID + c + j], vr[c + j] * sexp);
    if ((lane & 3) == 0) atomicAdd(&z[(size_t)di * NHEAD + (lane >> 2)], sexp);
  }
}

// h_attn = wV / (z + 1e-6), per head
__global__ void hattn_kernel(const float* __restrict__ wV, const float* __restrict__ z,
                             float* __restrict__ out, int total) {
  for (int i = blockIdx.x * blockDim.x + threadIdx.x; i < total;
       i += gridDim.x * blockDim.x) {
    int n = i >> 7, col = i & 127;
    out[i] = wV[i] / (z[n * NHEAD + (col >> 4)] + 1e-6f);
  }
}

// ---------------------------------------------------------------------------
// Fused: y1 = LN(X + P; g1,b1); t2 = relu(y1@W1 + B1);
//        t3 = y1 + (t2@W2 + B2); Out = LN(t3; g2,b2)
// 256 threads = 8 waves, 16-row tile per wave. Both FFN weights async-staged
// in LDS (f16); per-wave f32 y1 tile and f16 t2 tile in LDS. ~260KB LDS.
// LN: 2 lanes per row (64 cols each) + shfl_xor(16) combine.
// ---------------------------------------------------------------------------
__global__ void fused_ln_ffn_ln_kernel(
    const float* __restrict__ X, const float* __restrict__ P,
    const _Float16* __restrict__ W1p, const float* __restrict__ B1,
    const _Float16* __restrict__ W2p, const float* __restrict__ B2,
    const float* __restrict__ g1, const float* __restrict__ b1,
    const float* __restrict__ g2, const float* __restrict__ b2,
    float* __restrict__ Out, int numTiles) {
  extern __shared__ char smem[];
  _Float16* w1l = (_Float16*)smem;            // 128*256 f16
  _Float16* w2l = w1l + 128 * 256;            // 256*128 f16
  float* b1l  = (float*)(w2l + 256 * 128);    // 256
  float* b2l  = b1l + 256;                    // 128
  float* g1l  = b2l + 128;
  float* bb1l = g1l + 128;
  float* g2l  = bb1l + 128;
  float* bb2l = g2l + 128;
  float* tiles = bb2l + 128;
  const int nwaves = blockDim.x >> 5;
  float*     a1 = tiles + (threadIdx.x >> 5) * (16 * 128);          // f32 y1 tile
  _Float16* t2b = (_Float16*)(tiles + nwaves * 16 * 128);
  _Float16*  t2 = t2b + (threadIdx.x >> 5) * (16 * 256);            // f16 relu tile

  {  // async-stage both FFN weights; params via normal loads
    for (int i = threadIdx.x; i < (128 * 256 * 2) / 16; i += blockDim.x)
      async_ld_b128((char*)w1l + i * 16, (const char*)W1p + i * 16);
    for (int i = threadIdx.x; i < (256 * 128 * 2) / 16; i += blockDim.x)
      async_ld_b128((char*)w2l + i * 16, (const char*)W2p + i * 16);
    for (int i = threadIdx.x; i < 256; i += blockDim.x) b1l[i] = B1[i];
    for (int i = threadIdx.x; i < 128; i += blockDim.x) {
      b2l[i] = B2[i]; g1l[i] = g1[i]; bb1l[i] = b1[i]; g2l[i] = g2[i]; bb2l[i] = b2[i];
    }
  }
  wait_async0();
  __syncthreads();

  const int lane = threadIdx.x & 31;
  const int tile = blockIdx.x * nwaves + (threadIdx.x >> 5);
  if (tile >= numTiles) return;
  const size_t row0 = (size_t)tile * 16;

  // ---- stage 1: residual add + LN1 into a1 ----
  for (int idx = lane; idx < 16 * 128; idx += 32) {
    size_t r = row0 + (idx >> 7);
    int cc = idx & 127;
    a1[idx] = X[r * 128 + cc] + P[r * 128 + cc];
  }
  {
    const int r = lane & 15, ch = (lane >> 4) * 64;
    float mu = 0.0f, sq = 0.0f;
    for (int cc = ch; cc < ch + 64; ++cc) { float v = a1[r * 128 + cc]; mu += v; sq += v * v; }
    mu += __shfl_xor(mu, 16, 32);
    sq += __shfl_xor(sq, 16, 32);
    mu *= (1.0f / 128.0f);
    float rs = rsqrtf(sq * (1.0f / 128.0f) - mu * mu + 1e-5f);
    for (int cc = ch; cc < ch + 64; ++cc)
      a1[r * 128 + cc] = (a1[r * 128 + cc] - mu) * rs * g1l[cc] + bb1l[cc];
  }

  const int m = lane & 15, half = lane >> 4;

  // ---- stage 2: GEMM1 (128 -> 256) + bias + relu -> t2 (f16) ----
  v16h af[4];
#pragma unroll
  for (int kt = 0; kt < 4; ++kt) {
#pragma unroll
    for (int i = 0; i < 8; ++i) {
      int k = kt * 32 + a_kmap(i, half);
      af[kt][2 * i]     = (_Float16)a1[m * 128 + k];
      af[kt][2 * i + 1] = (_Float16)a1[m * 128 + k + 1];
    }
  }
#pragma unroll
  for (int nt = 0; nt < 16; ++nt) {
    v16h bf[4];
#pragma unroll
    for (int kt = 0; kt < 4; ++kt)
      __builtin_memcpy(&bf[kt], w1l + ((((nt * 4 + kt) * 32) + lane) << 4), sizeof(v16h));
    v8f acc = {};
#pragma unroll
    for (int kt = 0; kt < 4; ++kt) acc = wmma16(af[kt], bf[kt], acc);
    const int n = nt * 16 + m;
    const float bv = b1l[n];
#pragma unroll
    for (int j = 0; j < 8; ++j)
      t2[(j + 8 * half) * 256 + n] = (_Float16)fmaxf(acc[j] + bv, 0.0f);
  }

  // ---- stage 3: GEMM2 (256 -> 128) + bias + residual -> a1 ----
  v16h a2[8];
#pragma unroll
  for (int kt = 0; kt < 8; ++kt) {
#pragma unroll
    for (int i = 0; i < 8; ++i) {
      int k = kt * 32 + a_kmap(i, half);
      a2[kt][2 * i]     = t2[m * 256 + k];
      a2[kt][2 * i + 1] = t2[m * 256 + k + 1];
    }
  }
#pragma unroll
  for (int nt = 0; nt < 8; ++nt) {
    v16h bf[8];
#pragma unroll
    for (int kt = 0; kt < 8; ++kt)
      __builtin_memcpy(&bf[kt], w2l + ((((nt * 8 + kt) * 32) + lane) << 4), sizeof(v16h));
    v8f acc = {};
#pragma unroll
    for (int kt = 0; kt < 8; ++kt) acc = wmma16(a2[kt], bf[kt], acc);
    const int n = nt * 16 + m;
    const float bv = b2l[n];
#pragma unroll
    for (int j = 0; j < 8; ++j) {
      int mr = j + 8 * half;
      a1[mr * 128 + n] = acc[j] + bv + a1[mr * 128 + n];
    }
  }

  // ---- stage 4: LN2 -> Out ----
  {
    const int r = lane & 15, ch = (lane >> 4) * 64;
    float mu = 0.0f, sq = 0.0f;
    for (int cc = ch; cc < ch + 64; ++cc) { float v = a1[r * 128 + cc]; mu += v; sq += v * v; }
    mu += __shfl_xor(mu, 16, 32);
    sq += __shfl_xor(sq, 16, 32);
    mu *= (1.0f / 128.0f);
    float rs = rsqrtf(sq * (1.0f / 128.0f) - mu * mu + 1e-5f);
    for (int cc = ch; cc < ch + 64; ++cc)
      Out[(row0 + r) * 128 + cc] = (a1[r * 128 + cc] - mu) * rs * g2l[cc] + bb2l[cc];
  }
}

// ---------------------------------------------------------------------------
extern "C" void kernel_launch(void* const* d_in, const int* in_sizes, int n_in,
                              void* d_out, int out_size, void* d_ws, size_t ws_size,
                              hipStream_t stream) {
  (void)n_in; (void)out_size; (void)ws_size;
  const float* node  = (const float*)d_in[0];
  const float* edge  = (const float*)d_in[1];
  const float* wq    = (const float*)d_in[2];
  const float* wk    = (const float*)d_in[3];
  const float* wv    = (const float*)d_in[4];
  const float* we    = (const float*)d_in[5];
  const float* w_oh  = (const float*)d_in[6];
  const float* b_oh  = (const float*)d_in[7];
  const float* w_oe  = (const float*)d_in[8];
  const float* b_oe  = (const float*)d_in[9];
  const float* g1h   = (const float*)d_in[10];
  const float* b1h   = (const float*)d_in[11];
  const float* g1e   = (const float*)d_in[12];
  const float* b1e   = (const float*)d_in[13];
  const float* w_fh1 = (const float*)d_in[14];
  const float* b_fh1 = (const float*)d_in[15];
  const float* w_fh2 = (const float*)d_in[16];
  const float* b_fh2 = (const float*)d_in[17];
  const float* w_fe1 = (const float*)d_in[18];
  const float* b_fe1 = (const float*)d_in[19];
  const float* w_fe2 = (const float*)d_in[20];
  const float* b_fe2 = (const float*)d_in[21];
  const float* g2h   = (const float*)d_in[22];
  const float* b2h   = (const float*)d_in[23];
  const float* g2e   = (const float*)d_in[24];
  const float* b2e   = (const float*)d_in[25];
  const int*   src   = (const int*)d_in[26];
  const int*   dst   = (const int*)d_in[27];

  const int N = in_sizes[0] / HID;
  const int E = in_sizes[1] / HID;
  float* out_h = (float*)d_out;
  float* out_e = out_h + (size_t)N * HID;

  // workspace carve-out
  char* p = (char*)d_ws;
  auto alloc = [&](size_t bytes) -> char* {
    char* r = p;
    p += (bytes + 255) & ~(size_t)255;
    return r;
  };
  _Float16* wqP   = (_Float16*)alloc(128 * 128 * sizeof(_Float16));
  _Float16* wkP   = (_Float16*)alloc(128 * 128 * sizeof(_Float16));
  _Float16* wvP   = (_Float16*)alloc(128 * 128 * sizeof(_Float16));
  _Float16* weP   = (_Float16*)alloc(128 * 128 * sizeof(_Float16));
  _Float16* wohP  = (_Float16*)alloc(128 * 128 * sizeof(_Float16));
  _Float16* woeP  = (_Float16*)alloc(128 * 128 * sizeof(_Float16));
  _Float16* wfh1P = (_Float16*)alloc(128 * 256 * sizeof(_Float16));
  _Float16* wfh2P = (_Float16*)alloc(256 * 128 * sizeof(_Float16));
  _Float16* wfe1P = (_Float16*)alloc(128 * 256 * sizeof(_Float16));
  _Float16* wfe2P = (_Float16*)alloc(256 * 128 * sizeof(_Float16));
  float* Qb  = (float*)alloc((size_t)N * HID * 4);
  float* Kb  = (float*)alloc((size_t)N * HID * 4);
  float* Vb  = (float*)alloc((size_t)N * HID * 4);
  float* wVb = (float*)alloc((size_t)N * HID * 4);
  float* zb  = (float*)alloc((size_t)N * NHEAD * 4);
  float* peb = (float*)alloc((size_t)E * HID * 4);  // pe -> e_attn -> e_proj (in place)

  // 1) prepack weights to f16 fragment layout
  prepack_kernel<<<64, 256, 0, stream>>>(wq, wqP, 128, 128);
  prepack_kernel<<<64, 256, 0, stream>>>(wk, wkP, 128, 128);
  prepack_kernel<<<64, 256, 0, stream>>>(wv, wvP, 128, 128);
  prepack_kernel<<<64, 256, 0, stream>>>(we, weP, 128, 128);
  prepack_kernel<<<64, 256, 0, stream>>>(w_oh, wohP, 128, 128);
  prepack_kernel<<<64, 256, 0, stream>>>(w_oe, woeP, 128, 128);
  prepack_kernel<<<64, 256, 0, stream>>>(w_fh1, wfh1P, 128, 256);
  prepack_kernel<<<64, 256, 0, stream>>>(w_fh2, wfh2P, 256, 128);
  prepack_kernel<<<64, 256, 0, stream>>>(w_fe1, wfe1P, 128, 256);
  prepack_kernel<<<64, 256, 0, stream>>>(w_fe2, wfe2P, 256, 128);

  const int tilesN = N / 16, tilesE = E / 16;
  const int wgN = (tilesN + 7) / 8, wgE = (tilesE + 7) / 8;
  // gemm LDS: weights f16 + bias + 8 per-wave f32 A tiles
  const size_t smemG = (size_t)128 * 128 * sizeof(_Float16) + 128 * sizeof(float)
                       + 8 * (16 * 128) * sizeof(float);

  // 2) Q,K,V and pe GEMMs
  gemm_kernel<128, 128, false, false><<<wgN, 256, smemG, stream>>>(node, wqP, nullptr, Qb, tilesN);
  gemm_kernel<128, 128, false, false><<<wgN, 256, smemG, stream>>>(node, wkP, nullptr, Kb, tilesN);
  gemm_kernel<128, 128, false, false><<<wgN, 256, smemG, stream>>>(node, wvP, nullptr, Vb, tilesN);
  gemm_kernel<128, 128, false, false><<<wgE, 256, smemG, stream>>>(edge, weP, nullptr, peb, tilesE);

  // 3) segment-sum accumulators
  hipMemsetAsync(wVb, 0, (size_t)N * HID * 4, stream);
  hipMemsetAsync(zb, 0, (size_t)N * NHEAD * 4, stream);
  edge_attn_kernel<<<2048, 256, 0, stream>>>(Qb, Kb, Vb, peb, src, dst, wVb, zb, E);

  // 4) h_attn = wV/(z+1e-6) (into Qb, which is now dead)
  hattn_kernel<<<1024, 256, 0, stream>>>(wVb, zb, Qb, N * HID);

  // 5) output projections (in place)
  gemm_kernel<128, 128, true, false><<<wgN, 256, smemG, stream>>>(Qb, wohP, b_oh, Qb, tilesN);
  gemm_kernel<128, 128, true, false><<<wgE, 256, smemG, stream>>>(peb, woeP, b_oe, peb, tilesE);

  // 6) fused LN + FFN + LN for both paths, writing final outputs
  const size_t smemF = (size_t)(128 * 256 + 256 * 128) * sizeof(_Float16)  // weights
                       + (256 + 5 * 128) * sizeof(float)                   // bias/LN params
                       + 8 * (16 * 128) * sizeof(float)                    // y1 tiles
                       + 8 * (16 * 256) * sizeof(_Float16);                // relu tiles
  fused_ln_ffn_ln_kernel<<<wgN, 256, smemF, stream>>>(node, Qb, wfh1P, b_fh1, wfh2P, b_fh2,
                                                      g1h, b1h, g2h, b2h, out_h, tilesN);
  fused_ln_ffn_ln_kernel<<<wgE, 256, smemF, stream>>>(edge, peb, wfe1P, b_fe1, wfe2P, b_fe2,
                                                      g1e, b1e, g2e, b2e, out_e, tilesE);
}
